// Encoder_79740362817816
// MI455X (gfx1250) — matrix operand order repaired and checked by reading, not compiled, not run on hardware
//
#include <hip/hip_runtime.h>
#include <math.h>

// ---------------------------------------------------------------------------
// B=8 V=512 T=12 F0=4 F1=64 G=2 K=2 H=4 DK=DV=16 DINNER=128
// ALPHA=0.2 ALPHA_CGAT=512.0 NEG=-9e15
// ---------------------------------------------------------------------------

typedef __attribute__((ext_vector_type(16))) __bf16 v16bf;
typedef __attribute__((ext_vector_type(8)))  __bf16 v8bf_t;
typedef __attribute__((ext_vector_type(8)))  float  v8f;
typedef __attribute__((ext_vector_type(4)))  int    v4i;

#if defined(__has_builtin)
#if __has_builtin(__builtin_amdgcn_global_load_async_to_lds_b128) && \
    __has_builtin(__builtin_amdgcn_s_wait_asynccnt)
#define USE_ASYNC_COPY 1
#endif
#endif

// ------------------------- WMMA fragment loaders (bf16) --------------------
// A 16x32 bf16 (ISA 7.12.2): lane={half,m}, elem e -> K = (e<8?e:e+8)+8*half.
// Two contiguous 8-elem (16B) loads per lane.
__device__ __forceinline__ v16bf load_a_bf(const __bf16* __restrict__ A, int lda,
                                           int row0, int k0, int lane) {
  int m = lane & 15, half = lane >> 4;
  const __bf16* ap = A + (size_t)(row0 + m) * lda + k0 + (half << 3);
  v8bf_t lo = *(const v8bf_t*)ap;
  v8bf_t hi = *(const v8bf_t*)(ap + 16);
  v16bf a;
#pragma unroll
  for (int e = 0; e < 8; ++e) { a[e] = lo[e]; a[e + 8] = hi[e]; }
  return a;
}

// B 32x16 bf16: lane={half,n}, elem e -> K = (e<8?e:e+8)+8*half (matches A).
// Source stored as [n][k] row-major with leading dim ldb.
__device__ __forceinline__ v16bf load_b_bf(const __bf16* __restrict__ Bm, int ldb,
                                           int col0, int k0, int lane) {
  int n = lane & 15, half = lane >> 4;
  const __bf16* bp = Bm + (size_t)(col0 + n) * ldb + k0 + (half << 3);
  v8bf_t lo = *(const v8bf_t*)bp;
  v8bf_t hi = *(const v8bf_t*)(bp + 16);
  v16bf b;
#pragma unroll
  for (int e = 0; e < 8; ++e) { b[e] = lo[e]; b[e + 8] = hi[e]; }
  return b;
}

// ------------------------------ small kernels ------------------------------
__global__ void zero_kernel(float* __restrict__ p, int n) {
  int i = blockIdx.x * 256 + threadIdx.x;
  if (i < n) p[i] = 0.0f;
}

__global__ void cvt_bf16_kernel(const float* __restrict__ src, __bf16* __restrict__ dst, int n) {
  int i = blockIdx.x * 256 + threadIdx.x;
  if (i < n) dst[i] = (__bf16)src[i];
}

// cl[g,b,v,k] = softmax_k( xv(b,v,:48) . cW[g,k,:48] )
__global__ void cl_kernel(const float* __restrict__ x, const float* __restrict__ cW,
                          float* __restrict__ cl) {
  int idx = blockIdx.x * 256 + threadIdx.x;          // < G*B*V = 8192
  int g = idx / (8 * 512);
  int bv = idx % (8 * 512);
  const float* xp = x + (size_t)bv * 48;
  const float* w0 = cW + (size_t)(g * 2 + 0) * 48;
  const float* w1 = cW + (size_t)(g * 2 + 1) * 48;
  float l0 = 0.f, l1 = 0.f;
  for (int c = 0; c < 48; ++c) { float xv = xp[c]; l0 += xv * w0[c]; l1 += xv * w1[c]; }
  float mx = fmaxf(l0, l1);
  float e0 = expf(l0 - mx), e1 = expf(l1 - mx);
  float inv = 1.0f / (e0 + e1);
  cl[(size_t)idx * 2 + 0] = e0 * inv;
  cl[(size_t)idx * 2 + 1] = e1 * inv;
}

// h[b,v,t,f1] = leaky(x.Ww^T + Wb, 512); also emits transposed bf16
// haT[b][t*64+f1][u] for v = off+2u (this combo's atrous slice).
__global__ void cgat_h_kernel(const float* __restrict__ x, const float* __restrict__ Ww,
                              const float* __restrict__ Wb, int off,
                              float* __restrict__ h, __bf16* __restrict__ haT) {
  int idx = blockIdx.x * 256 + threadIdx.x;          // < B*V*T*64
  int f1 = idx & 63;
  int r  = idx >> 6;                                  // (b,v,t) flat
  const float* xp = x + (size_t)r * 4;
  const float* w  = Ww + (size_t)f1 * 4;
  float acc = Wb[f1] + xp[0] * w[0] + xp[1] * w[1] + xp[2] * w[2] + xp[3] * w[3];
  acc = acc >= 0.f ? acc : 512.0f * acc;
  h[idx] = acc;
  int t = r % 12; int v = (r / 12) % 512; int b = r / (12 * 512);
  if (((v ^ off) & 1) == 0) {
    int u = (v - off) >> 1;                           // 0..255
    haT[((size_t)b * 768 + t * 64 + f1) * 256 + u] = (__bf16)acc;
  }
}

// ht[v,f] = (1/8) sum_{b,t} h[b,v,t,f] * wt[t]
__global__ void cgat_ht_kernel(const float* __restrict__ h, const float* __restrict__ wt,
                               float* __restrict__ ht) {
  int idx = blockIdx.x * 256 + threadIdx.x;          // < V*64 = 32768
  int f = idx & 63, v = idx >> 6;
  float acc = 0.f;
  for (int b = 0; b < 8; ++b)
    for (int t = 0; t < 12; ++t)
      acc += h[(((size_t)b * 512 + v) * 12 + t) * 64 + f] * wt[t];
  ht[idx] = acc * 0.125f;
}

// s1[v] = ht[v,:] . aw[64:128] ;  s2[u] = ht[off+2u,:] . aw[0:64]
__global__ void cgat_s_kernel(const float* __restrict__ ht, const float* __restrict__ aw,
                              int off, float* __restrict__ s1, float* __restrict__ s2) {
  int tid = blockIdx.x * 256 + threadIdx.x;
  if (tid < 512) {
    const float* hp = ht + (size_t)tid * 64;
    float acc = 0.f;
    for (int f = 0; f < 64; ++f) acc += hp[f] * aw[64 + f];
    s1[tid] = acc;
  } else if (tid < 768) {
    int u = tid - 512;
    const float* hp = ht + (size_t)(off + 2 * u) * 64;
    float acc = 0.f;
    for (int f = 0; f < 64; ++f) acc += hp[f] * aw[f];
    s2[u] = acc;
  }
}

// attnw_bf[v,:] = softmax_u( adj ? leaky(s1[v]+s2[u],512) : -9e15 )  (bf16 out)
__global__ void cgat_softmax_kernel(const float* __restrict__ s1, const float* __restrict__ s2,
                                    const float* __restrict__ adj, int off,
                                    __bf16* __restrict__ attnw) {
  __shared__ float red[256];
  int v = blockIdx.x, u = threadIdx.x;
  float e = s1[v] + s2[u];
  e = e >= 0.f ? e : 512.0f * e;
  float a  = adj[(size_t)v * 512 + off + 2 * u];
  float sc = a > 0.f ? e : -9.0e15f;
  red[u] = sc; __syncthreads();
  for (int s = 128; s > 0; s >>= 1) { if (u < s) red[u] = fmaxf(red[u], red[u + s]); __syncthreads(); }
  float mx = red[0]; __syncthreads();
  float ex = expf(sc - mx);
  red[u] = ex; __syncthreads();
  for (int s = 128; s > 0; s >>= 1) { if (u < s) red[u] += red[u + s]; __syncthreads(); }
  attnw[(size_t)v * 256 + u] = (__bf16)(ex / red[0]);
}

// gc_act[b,v,:] += 0.5*cl[g,b,v,k] * leaky( attn(512x256) @ haT^T(256x768), 512 )
// one wave per 16x16 tile; K=256 -> 8 WMMA, all operands bf16 w/ b128 loads.
__global__ void cgat_apply_kernel(const __bf16* __restrict__ attnw, const __bf16* __restrict__ haT,
                                  const float* __restrict__ clbase, float* __restrict__ gc_act) {
  int wave = threadIdx.x >> 5, lane = threadIdx.x & 31;
  int b = blockIdx.y;
  int tile = blockIdx.x * 4 + wave;                  // 0..1535  (32 x 48)
  int tileM = tile / 48, tileN = tile % 48;
  int row0 = tileM << 4, col0 = tileN << 4;
  const __bf16* hB = haT + (size_t)b * 768 * 256;
  v8f c = {};
  for (int k0 = 0; k0 < 256; k0 += 32) {
    v16bf a  = load_a_bf(attnw, 256, row0, k0, lane);
    v16bf bm = load_b_bf(hB, 256, col0, k0, lane);
    c = __builtin_amdgcn_wmma_f32_16x16x32_bf16(false, a, false, bm, (short)0, c, false, false);
  }
  int n = lane & 15, half = lane >> 4;
  float* gB = gc_act + (size_t)b * 512 * 768;
#pragma unroll
  for (int r = 0; r < 8; ++r) {
    int m = (half << 3) + r;
    int v = row0 + m;
    float val = c[r];
    val = val >= 0.f ? val : 512.0f * val;           // ALPHA_CGAT
    float coeff = clbase[(size_t)(b * 512 + v) * 2] * 0.5f;   // cl/G
    gB[(size_t)v * 768 + col0 + n] += coeff * val;
  }
}

// D = act( A(MxK,bf16) @ W^T ), W bf16 NxK staged in LDS via async copy.
// Optional f32 and/or bf16 outputs. N*K <= 8192 elements (16 KB LDS).
__global__ void gemm_wt_kernel(const __bf16* __restrict__ A, const __bf16* __restrict__ W,
                               float* __restrict__ Df, __bf16* __restrict__ Db,
                               int M, int N, int K, float alpha, int act) {
  __shared__ __align__(16) __bf16 ldsW[8192];
  int total = N * K;
  for (int eofs = threadIdx.x * 8; eofs < total; eofs += 128 * 8) {
#ifdef USE_ASYNC_COPY
    __builtin_amdgcn_global_load_async_to_lds_b128(
        (__attribute__((address_space(1))) v4i*)(W + eofs),
        (__attribute__((address_space(3))) v4i*)(ldsW + eofs), 0, 0);
#else
    *(v8bf_t*)(ldsW + eofs) = *(const v8bf_t*)(W + eofs);
#endif
  }
#ifdef USE_ASYNC_COPY
  __builtin_amdgcn_s_wait_asynccnt(0);
#endif
  __syncthreads();

  int wave = threadIdx.x >> 5, lane = threadIdx.x & 31;
  int tile = blockIdx.x * 4 + wave;
  int tilesN = N >> 4;
  int tileM = tile / tilesN, tileN = tile % tilesN;
  int row0 = tileM << 4, col0 = tileN << 4;
  v8f c = {};
  for (int k0 = 0; k0 < K; k0 += 32) {
    v16bf a = load_a_bf(A, K, row0, k0, lane);
    v16bf b = load_b_bf(ldsW, K, col0, k0, lane);    // ds_load path
    c = __builtin_amdgcn_wmma_f32_16x16x32_bf16(false, a, false, b, (short)0, c, false, false);
  }
  int n = lane & 15, half = lane >> 4;
#pragma unroll
  for (int r = 0; r < 8; ++r) {
    int m = (half << 3) + r;
    float v = c[r];
    if (act) v = v >= 0.f ? v : alpha * v;
    size_t oidx = (size_t)(row0 + m) * N + col0 + n;
    if (Df) Df[oidx] = v;
    if (Db) Db[oidx] = (__bf16)v;
  }
}

// qk[b,t,f] = mean_v gc_act[b,v,t,f]
__global__ void qkmean_kernel(const float* __restrict__ gc, float* __restrict__ qk) {
  int idx = blockIdx.x * 256 + threadIdx.x;          // < 8*12*64 = 6144
  int f = idx & 63; int t = (idx >> 6) % 12; int b = idx / (12 * 64);
  float acc = 0.f;
  for (int v = 0; v < 512; ++v) acc += gc[(((size_t)b * 512 + v) * 12 + t) * 64 + f];
  qk[idx] = acc * (1.0f / 512.0f);
}

// q = qk @ wq^T, k = qk @ wk^T
__global__ void qkproj_kernel(const float* __restrict__ qk, const float* __restrict__ wq,
                              const float* __restrict__ wk, float* __restrict__ q,
                              float* __restrict__ kout) {
  int idx = blockIdx.x * 256 + threadIdx.x;          // < 8*12*64
  int o = idx & 63; int bt = idx >> 6;
  const float* qp  = qk + (size_t)bt * 64;
  const float* wqp = wq + (size_t)o * 64;
  const float* wkp = wk + (size_t)o * 64;
  float aq = 0.f, ak = 0.f;
  for (int f = 0; f < 64; ++f) { float xv = qp[f]; aq += xv * wqp[f]; ak += xv * wkp[f]; }
  q[idx] = aq; kout[idx] = ak;
}

// sattn[b,h,qt,:] = softmax_kt( q.k / 4 )
__global__ void tattn_kernel(const float* __restrict__ q, const float* __restrict__ k,
                             float* __restrict__ sattn) {
  int idx = blockIdx.x * 128 + threadIdx.x;          // < 384 = B*H*T
  if (idx >= 384) return;
  int qt = idx % 12; int h = (idx / 12) % 4; int b = idx / 48;
  const float* qp = q + (size_t)(b * 12 + qt) * 64 + h * 16;
  float sc[12]; float mx = -1e30f;
  for (int kt = 0; kt < 12; ++kt) {
    const float* kp = k + (size_t)(b * 12 + kt) * 64 + h * 16;
    float a = 0.f;
    for (int d = 0; d < 16; ++d) a += qp[d] * kp[d];
    a *= 0.25f;                                      // 1/sqrt(DK)
    sc[kt] = a; mx = fmaxf(mx, a);
  }
  float s = 0.f;
  for (int kt = 0; kt < 12; ++kt) { sc[kt] = expf(sc[kt] - mx); s += sc[kt]; }
  float inv = 1.0f / s;
  for (int kt = 0; kt < 12; ++kt) sattn[(size_t)idx * 12 + kt] = sc[kt] * inv;
}

// out_bf[b,v,qt, d*4+h] = sum_kt sattn[b,h,qt,kt] * vv[b,v,kt, h*16+d]
__global__ void attnout_kernel(const float* __restrict__ sattn, const float* __restrict__ vv,
                               __bf16* __restrict__ out) {
  int idx = blockIdx.x * 256 + threadIdx.x;          // < B*V*T*64
  int od = idx & 63; int d = od >> 2; int h = od & 3;
  int r = idx >> 6; int qt = r % 12; int bv = r / 12; int v = bv % 512; int b = bv / 512;
  const float* ap = sattn + ((size_t)(b * 4 + h) * 12 + qt) * 12;
  const float* vp = vv + ((size_t)(b * 512 + v) * 12) * 64 + h * 16 + d;
  float acc = 0.f;
  for (int kt = 0; kt < 12; ++kt) acc += ap[kt] * vp[(size_t)kt * 64];
  out[idx] = (__bf16)acc;
}

// ------------------------------- launcher ----------------------------------
extern "C" void kernel_launch(void* const* d_in, const int* in_sizes, int n_in,
                              void* d_out, int out_size, void* d_ws, size_t ws_size,
                              hipStream_t stream) {
  (void)in_sizes; (void)n_in; (void)out_size; (void)ws_size;
  const float* x      = (const float*)d_in[0];
  const float* graphs = (const float*)d_in[1];
  const float* cW     = (const float*)d_in[2];
  const float* Ww[2]  = {(const float*)d_in[3], (const float*)d_in[7]};
  const float* Wb[2]  = {(const float*)d_in[4], (const float*)d_in[8]};
  const float* wt[2]  = {(const float*)d_in[5], (const float*)d_in[9]};
  const float* aw[2]  = {(const float*)d_in[6], (const float*)d_in[10]};
  const float* wq = (const float*)d_in[11];
  const float* wk = (const float*)d_in[12];
  const float* wv = (const float*)d_in[13];
  const float* fc = (const float*)d_in[14];
  const float* w1 = (const float*)d_in[15];
  const float* w2 = (const float*)d_in[16];

  const int NE = 8 * 512 * 12 * 64;                  // 3145728

  float* ws = (float*)d_ws;
  size_t o = 0;
  float* cl    = ws + o; o += 2 * 8 * 512 * 2;       // 16384
  float* htb   = ws + o; o += 512 * 64;              // 32768
  float* s1    = ws + o; o += 512;
  float* s2    = ws + o; o += 256;
  float* qk    = ws + o; o += 8 * 12 * 64;
  float* qb    = ws + o; o += 8 * 12 * 64;
  float* kb    = ws + o; o += 8 * 12 * 64;
  float* sattn = ws + o; o += 8 * 4 * 12 * 12;
  o = (o + 7) & ~(size_t)7;
  float* hbuf  = ws + o; o += NE;                    // h f32, later vv f32
  float* gcact = ws + o; o += NE;                    // f32 accumulator

  __bf16* bw = (__bf16*)(ws + o);
  size_t ob = 0;
  __bf16* attnw_bf   = bw + ob; ob += 512 * 256;
  __bf16* haT_bf     = bw + ob; ob += (size_t)8 * 768 * 256;
  __bf16* gcact_bf   = bw + ob; ob += NE;
  __bf16* attnout_bf = bw + ob; ob += NE;
  __bf16* fcout_bf   = bw + ob; ob += NE;
  __bf16* t1_bf      = bw + ob; ob += (size_t)NE * 2;
  __bf16* wv_bf      = bw + ob; ob += 64 * 64;
  __bf16* fc_bf      = bw + ob; ob += 64 * 64;
  __bf16* w1_bf      = bw + ob; ob += 128 * 64;
  __bf16* w2_bf      = bw + ob; ob += 64 * 128;

  cl_kernel<<<32, 256, 0, stream>>>(x, cW, cl);
  zero_kernel<<<NE / 256, 256, 0, stream>>>(gcact, NE);
  cvt_bf16_kernel<<<16, 256, 0, stream>>>(wv, wv_bf, 64 * 64);
  cvt_bf16_kernel<<<16, 256, 0, stream>>>(fc, fc_bf, 64 * 64);
  cvt_bf16_kernel<<<32, 256, 0, stream>>>(w1, w1_bf, 128 * 64);
  cvt_bf16_kernel<<<32, 256, 0, stream>>>(w2, w2_bf, 64 * 128);

  for (int g = 0; g < 2; ++g)
    for (int kc = 0; kc < 2; ++kc)
      for (int off = 0; off < 2; ++off) {
        int gk = g * 2 + kc;
        cgat_h_kernel<<<NE / 256, 256, 0, stream>>>(x, Ww[off] + (size_t)gk * 64 * 4,
                                                    Wb[off] + (size_t)gk * 64, off, hbuf, haT_bf);
        cgat_ht_kernel<<<128, 256, 0, stream>>>(hbuf, wt[off] + (size_t)gk * 12, htb);
        cgat_s_kernel<<<3, 256, 0, stream>>>(htb, aw[off] + (size_t)gk * 128, off, s1, s2);
        cgat_softmax_kernel<<<512, 256, 0, stream>>>(s1, s2, graphs + (size_t)g * 512 * 512,
                                                     off, attnw_bf);
        dim3 gr(384, 8);
        cgat_apply_kernel<<<gr, 128, 0, stream>>>(attnw_bf, haT_bf,
                                                  cl + (size_t)g * 8 * 512 * 2 + kc, gcact);
      }

  qkmean_kernel<<<24, 256, 0, stream>>>(gcact, qk);
  qkproj_kernel<<<24, 256, 0, stream>>>(qk, wq, wk, qb, kb);
  cvt_bf16_kernel<<<NE / 256, 256, 0, stream>>>(gcact, gcact_bf, NE);
  // vv(f32) = gc_act @ wv^T   (M=49152, N=64, K=64) -> hbuf (h is dead)
  gemm_wt_kernel<<<3072, 128, 0, stream>>>(gcact_bf, wv_bf, hbuf, nullptr,
                                           49152, 64, 64, 0.0f, 0);
  tattn_kernel<<<3, 128, 0, stream>>>(qb, kb, sattn);
  attnout_kernel<<<NE / 256, 256, 0, stream>>>(sattn, hbuf, attnout_bf);
  // fcout(bf16) = leaky(attnout @ fc^T, .2)
  gemm_wt_kernel<<<3072, 128, 0, stream>>>(attnout_bf, fc_bf, nullptr, fcout_bf,
                                           49152, 64, 64, 0.2f, 1);
  // t1(bf16) = leaky(fcout @ w1^T, .2)   (N=128)
  gemm_wt_kernel<<<6144, 128, 0, stream>>>(fcout_bf, w1_bf, nullptr, t1_bf,
                                           49152, 128, 64, 0.2f, 1);
  // final(f32) = leaky(t1 @ w2^T, .2)    (K=128) -> d_out
  gemm_wt_kernel<<<3072, 128, 0, stream>>>(t1_bf, w2_bf, (float*)d_out, nullptr,
                                           49152, 64, 128, 0.2f, 1);
}
